// DeepseekV3MoE_72000831750712
// MI455X (gfx1250) — matrix operand order repaired and checked by reading, not compiled, not run on hardware
//
#include <hip/hip_runtime.h>
#include <hip/hip_bf16.h>

// ---------------- problem constants ----------------
constexpr int T    = 1024;   // tokens
constexpr int H    = 2048;   // hidden dim
constexpr int E    = 32;     // routed experts
constexpr int KSEL = 4;      // experts per token
constexpr int I_R  = 1408;   // routed intermediate
constexpr int I_S  = 2816;   // shared intermediate
constexpr float SCALE = 2.5f;

// ---------------- tiling ----------------
constexpr int BM = 128;      // token tile
constexpr int BN = 64;       // output-col tile
constexpr int BK = 32;       // k tile (bf16 wmma K)
constexpr int LROW = 40;     // LDS row stride (bf16), 80B -> conflict-free b128 (20*l mod 64 is a permutation)

typedef __attribute__((ext_vector_type(16))) __bf16 v16bf;
typedef __attribute__((ext_vector_type(8)))  float  v8f;
typedef __attribute__((ext_vector_type(4)))  float  nf4;   // clang ext vector for NT loads

// A-matrix fragment (16-bit 16x32): lane hi=0 -> K 0..15 ; hi=1 -> K 16..31 (contiguous)
__device__ inline v16bf frag_ldA(const __bf16* base, int row, int hi, int stride) {
    const __bf16* p = base + row * stride;
    union { uint4 q[2]; v16bf v; } r;
    r.q[0] = *reinterpret_cast<const uint4*>(p + hi * 16);
    r.q[1] = *reinterpret_cast<const uint4*>(p + hi * 16 + 8);
    return r.v;
}

// B-matrix fragment (16-bit 32x16): lane hi=0 -> K 0..15 ; hi=1 -> K 16..31 (contiguous)
__device__ inline v16bf frag_ldB(const __bf16* base, int row, int hi, int stride) {
    const __bf16* p = base + row * stride + hi * 16;
    union { uint4 q[2]; v16bf v; } r;
    r.q[0] = *reinterpret_cast<const uint4*>(p);
    r.q[1] = *reinterpret_cast<const uint4*>(p + 8);
    return r.v;
}

// async copy of 32 contiguous bytes: global (bf16) -> LDS, ASYNCcnt-tracked.
// the instruction offset is added to both the LDS and the global address
__device__ inline void async_cp32(const __bf16* gsrc, __bf16* ldst) {
    unsigned int  ldsa = (unsigned int)(unsigned long long)(uintptr_t)ldst; // low 32 bits = LDS offset
    unsigned long long ga = (unsigned long long)(uintptr_t)gsrc;
    asm volatile("global_load_async_to_lds_b128 %0, %1, off\n\t"
                 "global_load_async_to_lds_b128 %0, %1, off offset:16"
                 :: "v"(ldsa), "v"(ga) : "memory");
}

// pipeline barrier: drain *only* my async-LDS writes + my ds_stores, then split barrier.
// Crucially leaves LOADcnt alone so the register-staged weight loads for tile k+1
// stay in flight across the whole WMMA phase (hipcc's __syncthreads drains loadcnt too).
__device__ inline void pipeline_barrier() {
    asm volatile("s_wait_asynccnt 0x0\n\t"
                 "s_wait_dscnt 0x0\n\t"
                 "s_barrier_signal -1\n\t"
                 "s_barrier_wait -1" ::: "memory");
}

// ---------------- init: per-segment counters ----------------
__global__ void moe_init(int* __restrict__ cnt) {
    int i = threadIdx.x;
    if (i < E + 1) cnt[i] = (i == E) ? T : 0;   // shared segment has all tokens
}

// ---------------- x -> bf16 ----------------
__global__ void moe_cvt(const float* __restrict__ x, __bf16* __restrict__ xb, int total4) {
    int i = blockIdx.x * blockDim.x + threadIdx.x;
    if (i >= total4) return;
    float4 v = reinterpret_cast<const float4*>(x)[i];
    __bf16* d = xb + (long)i * 4;
    d[0] = (__bf16)v.x; d[1] = (__bf16)v.y; d[2] = (__bf16)v.z; d[3] = (__bf16)v.w;
}

// ---------------- gating + dispatch (one wave per token, lane == expert) ----------------
__global__ __launch_bounds__(256, 1)
void moe_gate(const float* __restrict__ x, const float* __restrict__ gw,
              const float* __restrict__ bias,
              int* __restrict__ cnt, int* __restrict__ seg_tok,
              int* __restrict__ seg_pair, float* __restrict__ seg_wt) {
    int tid = threadIdx.x, lane = tid & 31, wave = tid >> 5;
    int t = blockIdx.x * 8 + wave;

    const float4* xv = reinterpret_cast<const float4*>(x + (long)t * H);
    const float4* gv = reinterpret_cast<const float4*>(gw + (long)lane * H);
    float acc = 0.f;
    for (int h = 0; h < H / 4; ++h) {
        float4 a = xv[h], b = gv[h];
        acc += a.x * b.x + a.y * b.y + a.z * b.z + a.w * b.w;
    }
    float score = 1.0f / (1.0f + __expf(-acc));

    __shared__ float sc[8][32], cor[8][32];
    sc[wave][lane]  = score;
    cor[wave][lane] = score + bias[lane];
    __syncthreads();

    if (lane == 0) {
        float gs[8];
        for (int g = 0; g < 8; ++g) {
            float m1 = -1e30f, m2 = -1e30f;
            for (int j = 0; j < 4; ++j) {
                float v = cor[wave][g * 4 + j];
                if (v > m1) { m2 = m1; m1 = v; } else if (v > m2) m2 = v;
            }
            gs[g] = m1 + m2;
        }
        bool gsel[8] = {false,false,false,false,false,false,false,false};
        for (int r = 0; r < 4; ++r) {
            int bi = 0; float bv = -1e30f;
            for (int g = 0; g < 8; ++g) if (!gsel[g] && gs[g] > bv) { bv = gs[g]; bi = g; }
            gsel[bi] = true;
        }
        int idx[4]; float wsum = 0.f; bool used[32] = {false};
        for (int r = 0; r < 4; ++r) {
            int bi = 0; float bv = -1e30f;
            for (int e = 0; e < 32; ++e) {
                if (used[e] || !gsel[e >> 2]) continue;
                float v = cor[wave][e];
                if (v > bv) { bv = v; bi = e; }
            }
            used[bi] = true; idx[r] = bi; wsum += sc[wave][bi];
        }
        float inv = SCALE / (wsum + 1e-20f);
        for (int r = 0; r < 4; ++r) {
            int e = idx[r];
            int slot = atomicAdd(cnt + e, 1);
            seg_tok [e * T + slot] = t;
            seg_pair[e * T + slot] = t * KSEL + r;
            seg_wt  [e * T + slot] = sc[wave][e] * inv;
        }
        seg_tok [E * T + t] = t;
        seg_pair[E * T + t] = t;
        seg_wt  [E * T + t] = 1.0f;
    }
}

// ---------------- fused gate/up GEMM + SiLU  (A[pair, Idim] bf16 out) ----------------
// Software pipeline: B weights register-staged 1 tile ahead (fp32->bf16 in LDS, NT loads),
// A tiles copied with global_load_async_to_lds_b128 into ping-pong buffers.
__global__ __launch_bounds__(256, 1)
void moe_gateup(const __bf16* __restrict__ xb,
                const float* __restrict__ wg_base, const float* __restrict__ wu_base,
                long w_estride,
                const int* __restrict__ seg_tok, const int* __restrict__ seg_pair,
                const int* __restrict__ cnt,
                __bf16* __restrict__ Aout, int Idim, int seg0) {
    int seg = seg0 + blockIdx.z;
    int n   = cnt[seg];
    int m0  = blockIdx.y * BM;
    if (m0 >= n) return;
    int i0  = blockIdx.x * BN;

    const float* wg = wg_base + (long)blockIdx.z * w_estride;
    const float* wu = wu_base + (long)blockIdx.z * w_estride;
    const int* pairl = seg_pair + seg * T;
    const int* tokl  = seg_tok  + seg * T;

    __shared__ __bf16 As[2 * BM * LROW];
    __shared__ __bf16 Bg[2 * BN * LROW];
    __shared__ __bf16 Bu[2 * BN * LROW];

    int tid = threadIdx.x, lane = tid & 31, wave = tid >> 5;
    int wm = wave & 3, wn = wave >> 2;          // 4 x 2 wave grid
    int hi = lane >> 4, lr = lane & 15;

    // A gather: 2 threads per row, 32B each
    int arow = tid >> 1, apart = tid & 1;
    int aslot = m0 + arow;
    const __bf16* asrc = (aslot < n) ? (xb + (long)tokl[aslot] * H + apart * 16) : nullptr;
    __bf16* adst0 = As + arow * LROW + apart * 16;
    // B: 4 threads per row, 8 fp32 each (non-temporal: weights stream through L2 once)
    int brow = tid >> 2, bchunk = (tid & 3) * 8;
    const float* gsrc = wg + (long)(i0 + brow) * H + bchunk;
    const float* usrc = wu + (long)(i0 + brow) * H + bchunk;
    __bf16* bgdst0 = Bg + brow * LROW + bchunk;
    __bf16* budst0 = Bu + brow * LROW + bchunk;

    v8f cg[2][2] = {}; v8f cu[2][2] = {};
    nf4 rg0, rg1, ru0, ru1;                     // register-staged B tile

    // prologue: B(0) -> regs, A(0) -> Abuf[0] (async)
    {
        const nf4* s4 = reinterpret_cast<const nf4*>(gsrc);
        rg0 = __builtin_nontemporal_load(s4); rg1 = __builtin_nontemporal_load(s4 + 1);
        const nf4* t4 = reinterpret_cast<const nf4*>(usrc);
        ru0 = __builtin_nontemporal_load(t4); ru1 = __builtin_nontemporal_load(t4 + 1);
        if (asrc) async_cp32(asrc, adst0);
    }

    constexpr int KT = H / BK;                  // 64
    for (int kt = 0; kt < KT; ++kt) {
        int cur = kt & 1;
        // 1) convert + store B(kt) into Bbuf[cur]
        {
            __bf16* d = bgdst0 + cur * BN * LROW;
            d[0]=(__bf16)rg0.x; d[1]=(__bf16)rg0.y; d[2]=(__bf16)rg0.z; d[3]=(__bf16)rg0.w;
            d[4]=(__bf16)rg1.x; d[5]=(__bf16)rg1.y; d[6]=(__bf16)rg1.z; d[7]=(__bf16)rg1.w;
            __bf16* e2 = budst0 + cur * BN * LROW;
            e2[0]=(__bf16)ru0.x; e2[1]=(__bf16)ru0.y; e2[2]=(__bf16)ru0.z; e2[3]=(__bf16)ru0.w;
            e2[4]=(__bf16)ru1.x; e2[5]=(__bf16)ru1.y; e2[6]=(__bf16)ru1.z; e2[7]=(__bf16)ru1.w;
        }
        // 2) issue global loads for B(kt+1); prefetch B(kt+2)
        int knext = (kt + 1) * BK;
        if (kt + 1 < KT) {
            const nf4* s4 = reinterpret_cast<const nf4*>(gsrc + knext);
            rg0 = __builtin_nontemporal_load(s4); rg1 = __builtin_nontemporal_load(s4 + 1);
            const nf4* t4 = reinterpret_cast<const nf4*>(usrc + knext);
            ru0 = __builtin_nontemporal_load(t4); ru1 = __builtin_nontemporal_load(t4 + 1);
            if (kt + 2 < KT) {
                __builtin_prefetch(gsrc + knext + BK, 0, 1);
                __builtin_prefetch(usrc + knext + BK, 0, 1);
            }
        }
        // 3) drain my asyncs + ds_stores, barrier (LOADcnt stays in flight)
        pipeline_barrier();
        // 4) A(kt+1) async into the buffer nobody reads anymore
        if (kt + 1 < KT && asrc) async_cp32(asrc + knext, adst0 + (1 - cur) * BM * LROW);
        // 5) compute on buf[cur]
        const __bf16* a_ = As + cur * BM * LROW;
        const __bf16* g_ = Bg + cur * BN * LROW;
        const __bf16* u_ = Bu + cur * BN * LROW;
        v16bf af[2], bgf[2], buf2[2];
        #pragma unroll
        for (int im = 0; im < 2; ++im) af[im] = frag_ldA(a_, wm * 32 + im * 16 + lr, hi, LROW);
        #pragma unroll
        for (int in = 0; in < 2; ++in) {
            bgf[in]  = frag_ldB(g_, wn * 32 + in * 16 + lr, hi, LROW);
            buf2[in] = frag_ldB(u_, wn * 32 + in * 16 + lr, hi, LROW);
        }
        #pragma unroll
        for (int im = 0; im < 2; ++im)
            #pragma unroll
            for (int in = 0; in < 2; ++in) {
                cg[im][in] = __builtin_amdgcn_wmma_f32_16x16x32_bf16(
                    false, af[im], false, bgf[in], (short)0, cg[im][in], false, false);
                cu[im][in] = __builtin_amdgcn_wmma_f32_16x16x32_bf16(
                    false, af[im], false, buf2[in], (short)0, cu[im][in], false, false);
            }
    }

    // epilogue: a = silu(g) * u -> bf16, scattered by pair index
    #pragma unroll
    for (int im = 0; im < 2; ++im)
        #pragma unroll
        for (int in = 0; in < 2; ++in)
            #pragma unroll
            for (int v = 0; v < 8; ++v) {
                int m = wm * 32 + im * 16 + hi * 8 + v;
                int slot = m0 + m;
                if (slot < n) {
                    int pr = pairl[slot];
                    int i  = i0 + wn * 32 + in * 16 + lr;
                    float g = cg[im][in][v], u = cu[im][in][v];
                    float a = (g / (1.0f + __expf(-g))) * u;
                    Aout[(long)pr * Idim + i] = (__bf16)a;
                }
            }
}

// ---------------- down GEMM: y[t,h] (+)= wt * (A @ Wd^T) ----------------
template <bool ATOMIC>
__global__ __launch_bounds__(256, 1)
void moe_down(const __bf16* __restrict__ Abuf,
              const float* __restrict__ wd_base, long wd_estride,
              const int* __restrict__ seg_tok, const int* __restrict__ seg_pair,
              const float* __restrict__ seg_wt, const int* __restrict__ cnt,
              float* __restrict__ y, int Idim, int seg0) {
    int seg = seg0 + blockIdx.z;
    int n   = cnt[seg];
    int m0  = blockIdx.y * BM;
    if (m0 >= n) return;
    int h0  = blockIdx.x * BN;

    const float* wd  = wd_base + (long)blockIdx.z * wd_estride;
    const int* tokl  = seg_tok  + seg * T;
    const int* pairl = seg_pair + seg * T;
    const float* wtl = seg_wt   + seg * T;

    __shared__ __bf16 As[2 * BM * LROW];
    __shared__ __bf16 Bd[2 * BN * LROW];

    int tid = threadIdx.x, lane = tid & 31, wave = tid >> 5;
    int wm = wave & 3, wn = wave >> 2;
    int hi = lane >> 4, lr = lane & 15;

    int arow = tid >> 1, apart = tid & 1;
    int aslot = m0 + arow;
    const __bf16* asrc = (aslot < n) ? (Abuf + (long)pairl[aslot] * Idim + apart * 16) : nullptr;
    __bf16* adst0 = As + arow * LROW + apart * 16;
    int brow = tid >> 2, bchunk = (tid & 3) * 8;
    const float* bsrc = wd + (long)(h0 + brow) * Idim + bchunk;
    __bf16* bdst0 = Bd + brow * LROW + bchunk;

    v8f c[2][2] = {};
    nf4 rb0, rb1;

    {
        const nf4* s4 = reinterpret_cast<const nf4*>(bsrc);
        rb0 = __builtin_nontemporal_load(s4); rb1 = __builtin_nontemporal_load(s4 + 1);
        if (asrc) async_cp32(asrc, adst0);
    }

    const int KT = Idim / BK;
    for (int kt = 0; kt < KT; ++kt) {
        int cur = kt & 1;
        {
            __bf16* d = bdst0 + cur * BN * LROW;
            d[0]=(__bf16)rb0.x; d[1]=(__bf16)rb0.y; d[2]=(__bf16)rb0.z; d[3]=(__bf16)rb0.w;
            d[4]=(__bf16)rb1.x; d[5]=(__bf16)rb1.y; d[6]=(__bf16)rb1.z; d[7]=(__bf16)rb1.w;
        }
        int knext = (kt + 1) * BK;
        if (kt + 1 < KT) {
            const nf4* s4 = reinterpret_cast<const nf4*>(bsrc + knext);
            rb0 = __builtin_nontemporal_load(s4); rb1 = __builtin_nontemporal_load(s4 + 1);
            if (kt + 2 < KT) __builtin_prefetch(bsrc + knext + BK, 0, 1);
        }
        pipeline_barrier();
        if (kt + 1 < KT && asrc) async_cp32(asrc + knext, adst0 + (1 - cur) * BM * LROW);

        const __bf16* a_ = As + cur * BM * LROW;
        const __bf16* b_ = Bd + cur * BN * LROW;
        v16bf af[2], bf2[2];
        #pragma unroll
        for (int im = 0; im < 2; ++im) af[im] = frag_ldA(a_, wm * 32 + im * 16 + lr, hi, LROW);
        #pragma unroll
        for (int in = 0; in < 2; ++in) bf2[in] = frag_ldB(b_, wn * 32 + in * 16 + lr, hi, LROW);
        #pragma unroll
        for (int im = 0; im < 2; ++im)
            #pragma unroll
            for (int in = 0; in < 2; ++in)
                c[im][in] = __builtin_amdgcn_wmma_f32_16x16x32_bf16(
                    false, af[im], false, bf2[in], (short)0, c[im][in], false, false);
    }

    #pragma unroll
    for (int im = 0; im < 2; ++im)
        #pragma unroll
        for (int in = 0; in < 2; ++in)
            #pragma unroll
            for (int v = 0; v < 8; ++v) {
                int m = wm * 32 + im * 16 + hi * 8 + v;
                int slot = m0 + m;
                if (slot < n) {
                    int tt = tokl[slot];
                    float w = wtl[slot];
                    int h = h0 + wn * 32 + in * 16 + lr;
                    float val = w * c[im][in][v];
                    if (ATOMIC) atomicAdd(y + (long)tt * H + h, val);
                    else        y[(long)tt * H + h] = val;
                }
            }
}

// ---------------- launch ----------------
extern "C" void kernel_launch(void* const* d_in, const int* in_sizes, int n_in,
                              void* d_out, int out_size, void* d_ws, size_t ws_size,
                              hipStream_t stream) {
    (void)in_sizes; (void)n_in; (void)out_size; (void)ws_size;
    const float* x    = (const float*)d_in[0];
    const float* gw   = (const float*)d_in[1];
    const float* bias = (const float*)d_in[2];
    const float* wgt  = (const float*)d_in[3];
    const float* wup  = (const float*)d_in[4];
    const float* wdn  = (const float*)d_in[5];
    const float* swg  = (const float*)d_in[6];
    const float* swu  = (const float*)d_in[7];
    const float* swd  = (const float*)d_in[8];
    float* y = (float*)d_out;

    char* ws = (char*)d_ws;
    int*    cnt   = (int*)  (ws + 0);          // 33 ints
    int*    stok  = (int*)  (ws + 256);        // 33*1024 ints
    int*    spair = (int*)  (ws + 135424);     // 33*1024 ints
    float*  swt   = (float*)(ws + 270592);     // 33*1024 floats
    __bf16* xb    = (__bf16*)(ws + 405760);    // 1024*2048 bf16  (4 MB)
    __bf16* Ar    = (__bf16*)(ws + 4600064);   // 4096*1408 bf16  (11.5 MB)
    __bf16* Ash   = (__bf16*)(ws + 16134400);  // 1024*2816 bf16  (5.8 MB)

    moe_init<<<1, 64, 0, stream>>>(cnt);
    moe_cvt <<<(T * H / 4 + 255) / 256, 256, 0, stream>>>(x, xb, T * H / 4);
    moe_gate<<<T / 8, 256, 0, stream>>>(x, gw, bias, cnt, stok, spair, swt);

    moe_gateup<<<dim3(I_R / BN, T / BM, E), 256, 0, stream>>>(
        xb, wgt, wup, (long)I_R * H, stok, spair, cnt, Ar, I_R, 0);
    moe_gateup<<<dim3(I_S / BN, T / BM, 1), 256, 0, stream>>>(
        xb, swg, swu, 0, stok, spair, cnt, Ash, I_S, E);

    moe_down<false><<<dim3(H / BN, T / BM, 1), 256, 0, stream>>>(
        Ash, swd, 0, stok, spair, swt, cnt, y, I_S, E);
    moe_down<true><<<dim3(H / BN, T / BM, E), 256, 0, stream>>>(
        Ar, wdn, (long)H * I_R, stok, spair, swt, cnt, y, I_R, 0);
}
